// Block_74500502716421
// MI455X (gfx1250) — matrix operand order repaired and checked by reading, not compiled, not run on hardware
//
#include <hip/hip_runtime.h>
#include <hip/hip_bf16.h>

// ---------------------------------------------------------------------------
// Transformer block for MI455X (gfx1250): all GEMMs via v_wmma_f32_16x16x32_f16
// Register-blocked (2x2 / fused-QKV) so each loaded fragment feeds >=2 WMMAs.
// ---------------------------------------------------------------------------

typedef __attribute__((ext_vector_type(16))) _Float16 v16h;
typedef __attribute__((ext_vector_type(8)))  _Float16 v8h;
typedef __attribute__((ext_vector_type(8)))  float    v8f;

#define WMMA_F16(a, b, c) \
    __builtin_amdgcn_wmma_f32_16x16x32_f16(false, (a), false, (b), (short)0, (c), false, false)

#define B_SZ   512
#define T_SEQ  128
#define EMB    512
#define NH     8
#define HD     64
#define FF     2048
#define FCHUNK 512   // FFN processed in 4 chunks of 512 columns

// LDS strides (halves / floats), padded for bank-conflict-free 16-lane B128 access
#define H16_LD 520
#define QS_LD  72
#define VT_LD  136
#define SC_LD  132
#define AT_LD  136
// LDS layout (bytes): h16 133120 | qs 18432 | ks 18432 | vT 17408 | sc 67584 | att 34816
#define SMEM_BYTES 289792

// ---- A-fragment: 16x32 f16, src row-major [rows][ld], rows m0.., cols k0..k0+31
// lanes 0-15: row=lane, K {k0..k0+7, k0+16..k0+23}; lanes 16-31: K {+8, +24}
__device__ __forceinline__ v16h frag_ld_a(const _Float16* base, int ld, int row0, int k0) {
    int lane = threadIdx.x & 31;
    const _Float16* p = base + (size_t)(row0 + (lane & 15)) * ld + k0 + ((lane >> 4) << 3);
    v8h lo = *(const v8h*)p;
    v8h hi = *(const v8h*)(p + 16);
    v16h r;
#pragma unroll
    for (int j = 0; j < 8; ++j) { r[j] = lo[j]; r[j + 8] = hi[j]; }
    return r;
}

// ---- B-fragment: 32x16 f16 where B[k][n] = src[n][k] (src = transposed weights
// [N][K] row-major). lane = column n; lanes 0-15: K {k0..k0+7, k0+16..k0+23}; lanes 16-31: K {+8, +24}.
__device__ __forceinline__ v16h frag_ld_b(const _Float16* base, int ld, int n0, int k0) {
    int lane = threadIdx.x & 31;
    const _Float16* p = base + (size_t)(n0 + (lane & 15)) * ld + k0 + ((lane >> 4) << 3);
    v8h lo = *(const v8h*)p;
    v8h hi = *(const v8h*)(p + 16);
    v16h r;
#pragma unroll
    for (int j = 0; j < 8; ++j) { r[j] = lo[j]; r[j + 8] = hi[j]; }
    return r;
}

// ---- C/D fragment store helpers: VGPR r -> row = row0 + 8*(lane/16) + r, col = lane%16
__device__ __forceinline__ void cstore_f16(_Float16* base, int ld, int row0, int col0, v8f acc) {
    int lane = threadIdx.x & 31;
    int r0 = row0 + ((lane >> 4) << 3);
    int c  = col0 + (lane & 15);
#pragma unroll
    for (int j = 0; j < 8; ++j) base[(size_t)(r0 + j) * ld + c] = (_Float16)acc[j];
}

// store transposed: dst[col][row] (used to build v^T in LDS; contiguous per lane)
__device__ __forceinline__ void cstore_f16_T(_Float16* base, int ld, int row0, int col0, v8f acc) {
    int lane = threadIdx.x & 31;
    int s0 = row0 + ((lane >> 4) << 3);
    int d  = col0 + (lane & 15);
    _Float16* p = base + (size_t)d * ld + s0;
#pragma unroll
    for (int j = 0; j < 8; ++j) p[j] = (_Float16)acc[j];
}

// ---------------------------------------------------------------------------
// Weight convert + transpose: src f32 [R][C] (x nb batches) -> dst f16 [C][R]
// ---------------------------------------------------------------------------
__global__ void convt_kernel(const float* __restrict__ src, _Float16* __restrict__ dst,
                             int R, int C) {
    size_t bo = (size_t)blockIdx.y * R * C;
    int i = blockIdx.x * blockDim.x + threadIdx.x;
    if (i < R * C) {
        int r = i / C, c = i % C;
        dst[bo + (size_t)c * R + r] = (_Float16)src[bo + i];
    }
}

// ---------------------------------------------------------------------------
// Fused LN1 + QKV + causal attention, one workgroup (16 waves) per batch
// ---------------------------------------------------------------------------
__global__ __launch_bounds__(512)
void attn_kernel(const float* __restrict__ x,
                 const _Float16* __restrict__ wqt,  // [H][D][E]
                 const _Float16* __restrict__ wkt,
                 const _Float16* __restrict__ wvt,
                 const float* __restrict__ ln1_g, const float* __restrict__ ln1_b,
                 _Float16* __restrict__ o16)       // [B][T][E] heads concatenated
{
    extern __shared__ char smem[];
    _Float16* h16 = (_Float16*)smem;                      // [128][520]
    _Float16* qs  = h16 + 128 * H16_LD;                   // [128][72]
    _Float16* ks  = qs + 128 * QS_LD;                     // [128][72]
    _Float16* vT  = ks + 128 * QS_LD;                     // [64][136]
    float*    sc  = (float*)(vT + 64 * VT_LD);            // [128][132]
    _Float16* att = (_Float16*)(sc + 128 * SC_LD);        // [128][136]

    const int b    = blockIdx.x;
    const int tid  = threadIdx.x;
    const int wave = tid >> 5;
    const int lane = tid & 31;
    const int NW   = blockDim.x >> 5;   // 16 waves

    // ---------------- LN1 -> h16 (f16, LDS). One wave per row, coalesced.
    const float* xb = x + (size_t)b * T_SEQ * EMB;
    for (int r = wave; r < T_SEQ; r += NW) {
        const float* xr = xb + (size_t)r * EMB;
        float s = 0.f, s2 = 0.f;
        for (int e = lane; e < EMB; e += 32) { float v = xr[e]; s += v; s2 += v * v; }
#pragma unroll
        for (int m = 16; m >= 1; m >>= 1) { s += __shfl_xor(s, m, 32); s2 += __shfl_xor(s2, m, 32); }
        float mu   = s * (1.0f / EMB);
        float var  = s2 * (1.0f / EMB) - mu * mu;
        float rinv = rsqrtf(var + 1e-5f);
        _Float16* hr = h16 + (size_t)r * H16_LD;
        for (int e = lane; e < EMB; e += 32)
            hr[e] = (_Float16)(((xr[e] - mu) * rinv) * ln1_g[e] + ln1_b[e]);
    }
    __syncthreads();

    for (int h = 0; h < NH; ++h) {
        const _Float16* wqh = wqt + (size_t)h * HD * EMB;
        const _Float16* wkh = wkt + (size_t)h * HD * EMB;
        const _Float16* wvh = wvt + (size_t)h * HD * EMB;

        // ------------ fused QKV: 8 mtiles x 4 ntiles jobs; A-frag reused for q,k,v
        for (int job = wave; job < 32; job += NW) {
            int mt = job >> 2;
            int nt = job & 3;
            v8f aq = {}, ak = {}, av = {};
            for (int kk = 0; kk < EMB; kk += 32) {
                v16h a  = frag_ld_a(h16, H16_LD, mt * 16, kk);
                v16h bq = frag_ld_b(wqh, EMB, nt * 16, kk);
                v16h bk = frag_ld_b(wkh, EMB, nt * 16, kk);
                v16h bv = frag_ld_b(wvh, EMB, nt * 16, kk);
                aq = WMMA_F16(a, bq, aq);
                ak = WMMA_F16(a, bk, ak);
                av = WMMA_F16(a, bv, av);
            }
            cstore_f16  (qs, QS_LD, mt * 16, nt * 16, aq);
            cstore_f16  (ks, QS_LD, mt * 16, nt * 16, ak);
            cstore_f16_T(vT, VT_LD, mt * 16, nt * 16, av);
        }
        __syncthreads();

        // ------------ scores = (q k^T) * 1/sqrt(D); lower-triangle tiles only (36)
        for (int job = wave; job < 36; job += NW) {
            int mt = 0, rem = job;
            while (rem > mt) { rem -= (mt + 1); ++mt; }
            int nt = rem;
            v8f acc = {};
            for (int kk = 0; kk < HD; kk += 32) {
                v16h a  = frag_ld_a(qs, QS_LD, mt * 16, kk);
                v16h bb = frag_ld_b(ks, QS_LD, nt * 16, kk);   // B = k^T
                acc = WMMA_F16(a, bb, acc);
            }
            int r0 = mt * 16 + ((lane >> 4) << 3);
            int c  = nt * 16 + (lane & 15);
#pragma unroll
            for (int j = 0; j < 8; ++j) sc[(size_t)(r0 + j) * SC_LD + c] = acc[j] * 0.125f;
        }
        __syncthreads();

        // ------------ causal softmax per row (one wave per row), write f16 attn
        for (int r = wave; r < T_SEQ; r += NW) {
            float*    srow = sc  + (size_t)r * SC_LD;
            _Float16* arow = att + (size_t)r * AT_LD;
            float mx = -3.0e38f;
            for (int c = lane; c <= r; c += 32) mx = fmaxf(mx, srow[c]);
#pragma unroll
            for (int m = 16; m >= 1; m >>= 1) mx = fmaxf(mx, __shfl_xor(mx, m, 32));
            float sum = 0.f;
            for (int c = lane; c <= r; c += 32) {
                float e = __expf(srow[c] - mx);
                srow[c] = e;
                sum += e;
            }
#pragma unroll
            for (int m = 16; m >= 1; m >>= 1) sum += __shfl_xor(sum, m, 32);
            float inv = 1.0f / sum;
            for (int c = lane; c < T_SEQ; c += 32)
                arow[c] = (c <= r) ? (_Float16)(srow[c] * inv) : (_Float16)0.f;
        }
        __syncthreads();

        // ------------ o = attn @ v : 8 mtiles x (2 pairs of d-tiles); A reused 2x
        _Float16* ob = o16 + (size_t)b * T_SEQ * EMB + h * HD;
        for (int job = wave; job < 16; job += NW) {
            int mt = job >> 1;
            int np = job & 1;          // d-tile pair: {2*np, 2*np+1}
            v8f acc0 = {}, acc1 = {};
            for (int kk = 0; kk < T_SEQ; kk += 32) {
                v16h a  = frag_ld_a(att, AT_LD, mt * 16, kk);
                v16h b0 = frag_ld_b(vT,  VT_LD, (np * 2) * 16, kk);
                v16h b1 = frag_ld_b(vT,  VT_LD, (np * 2 + 1) * 16, kk);
                acc0 = WMMA_F16(a, b0, acc0);
                acc1 = WMMA_F16(a, b1, acc1);
            }
            cstore_f16(ob, EMB, mt * 16, (np * 2) * 16, acc0);
            cstore_f16(ob, EMB, mt * 16, (np * 2 + 1) * 16, acc1);
        }
        __syncthreads();
    }
}

// ---------------------------------------------------------------------------
// Generic 2x2-blocked wave-tile GEMM body (M=128 rows, N=512 cols, K=512)
//   decode: per batch 4 mblk x 16 nblk = 64 jobs; 8 waves/block
// ---------------------------------------------------------------------------
#define GEMM_DECODE()                                   \
    int idx  = blockIdx.x * 8 + (threadIdx.x >> 5);     \
    int lane = threadIdx.x & 31;                        \
    int b    = idx >> 6;                                \
    int r_   = idx & 63;                                \
    int mt0  = (r_ >> 4) * 2;                           \
    int nt0  = (r_ & 15) * 2;

// ---------------------------------------------------------------------------
// x1 = x + o16 @ proj_w + proj_b   (f32 -> d_out)
// ---------------------------------------------------------------------------
__global__ __launch_bounds__(256)
void proj_kernel(const _Float16* __restrict__ o16, const _Float16* __restrict__ wt, // [E][E] transposed
                 const float* __restrict__ bias, const float* __restrict__ x,
                 float* __restrict__ out)
{
    GEMM_DECODE();
    const _Float16* A = o16 + (size_t)b * T_SEQ * EMB;
    v8f a00 = {}, a01 = {}, a10 = {}, a11 = {};
    for (int kk = 0; kk < EMB; kk += 32) {
        if (kk + 32 < EMB) __builtin_prefetch(wt + (size_t)nt0 * 16 * EMB + kk + 32, 0, 1);
        v16h fa0 = frag_ld_a(A, EMB, mt0 * 16, kk);
        v16h fa1 = frag_ld_a(A, EMB, (mt0 + 1) * 16, kk);
        v16h fb0 = frag_ld_b(wt, EMB, nt0 * 16, kk);
        v16h fb1 = frag_ld_b(wt, EMB, (nt0 + 1) * 16, kk);
        a00 = WMMA_F16(fa0, fb0, a00);
        a01 = WMMA_F16(fa0, fb1, a01);
        a10 = WMMA_F16(fa1, fb0, a10);
        a11 = WMMA_F16(fa1, fb1, a11);
    }
    const float* xb = x   + (size_t)b * T_SEQ * EMB;
    float*       ob = out + (size_t)b * T_SEQ * EMB;
#pragma unroll
    for (int q = 0; q < 4; ++q) {
        v8f acc = (q == 0) ? a00 : (q == 1) ? a01 : (q == 2) ? a10 : a11;
        int r0 = (mt0 + (q >> 1)) * 16 + ((lane >> 4) << 3);
        int c  = (nt0 + (q & 1)) * 16 + (lane & 15);
        float bv = bias[c];
#pragma unroll
        for (int j = 0; j < 8; ++j)
            ob[(size_t)(r0 + j) * EMB + c] = xb[(size_t)(r0 + j) * EMB + c] + acc[j] + bv;
    }
}

// ---------------------------------------------------------------------------
// LayerNorm (f32 in -> f16 out), one wave per row over B*T rows
// ---------------------------------------------------------------------------
__global__ __launch_bounds__(256)
void ln_kernel(const float* __restrict__ in, const float* __restrict__ g,
               const float* __restrict__ bta, _Float16* __restrict__ out)
{
    int row  = blockIdx.x * 8 + (threadIdx.x >> 5);
    int lane = threadIdx.x & 31;
    const float* xr = in + (size_t)row * EMB;
    float s = 0.f, s2 = 0.f;
    for (int e = lane; e < EMB; e += 32) { float v = xr[e]; s += v; s2 += v * v; }
#pragma unroll
    for (int m = 16; m >= 1; m >>= 1) { s += __shfl_xor(s, m, 32); s2 += __shfl_xor(s2, m, 32); }
    float mu   = s * (1.0f / EMB);
    float var  = s2 * (1.0f / EMB) - mu * mu;
    float rinv = rsqrtf(var + 1e-5f);
    _Float16* orow = out + (size_t)row * EMB;
    for (int e = lane; e < EMB; e += 32)
        orow[e] = (_Float16)(((xr[e] - mu) * rinv) * g[e] + bta[e]);
}

// ---------------------------------------------------------------------------
// FFN1 chunk: u = relu(h2 @ ff1_w[:, nOff:nOff+512] + b1[nOff:])  -> f16
// ---------------------------------------------------------------------------
__global__ __launch_bounds__(256)
void ffn1_kernel(const _Float16* __restrict__ h2, const _Float16* __restrict__ w1t, // [FF][E]
                 const float* __restrict__ b1, _Float16* __restrict__ u, int nOff)
{
    GEMM_DECODE();
    const _Float16* A  = h2  + (size_t)b * T_SEQ * EMB;
    const _Float16* Bt = w1t + (size_t)nOff * EMB;
    v8f a00 = {}, a01 = {}, a10 = {}, a11 = {};
    for (int kk = 0; kk < EMB; kk += 32) {
        if (kk + 32 < EMB) __builtin_prefetch(Bt + (size_t)nt0 * 16 * EMB + kk + 32, 0, 1);
        v16h fa0 = frag_ld_a(A, EMB, mt0 * 16, kk);
        v16h fa1 = frag_ld_a(A, EMB, (mt0 + 1) * 16, kk);
        v16h fb0 = frag_ld_b(Bt, EMB, nt0 * 16, kk);
        v16h fb1 = frag_ld_b(Bt, EMB, (nt0 + 1) * 16, kk);
        a00 = WMMA_F16(fa0, fb0, a00);
        a01 = WMMA_F16(fa0, fb1, a01);
        a10 = WMMA_F16(fa1, fb0, a10);
        a11 = WMMA_F16(fa1, fb1, a11);
    }
    _Float16* ub = u + (size_t)b * T_SEQ * FCHUNK;
#pragma unroll
    for (int q = 0; q < 4; ++q) {
        v8f acc = (q == 0) ? a00 : (q == 1) ? a01 : (q == 2) ? a10 : a11;
        int r0 = (mt0 + (q >> 1)) * 16 + ((lane >> 4) << 3);
        int c  = (nt0 + (q & 1)) * 16 + (lane & 15);
        float bv = b1[nOff + c];
#pragma unroll
        for (int j = 0; j < 8; ++j)
            ub[(size_t)(r0 + j) * FCHUNK + c] = (_Float16)fmaxf(acc[j] + bv, 0.f);
    }
}

// ---------------------------------------------------------------------------
// FFN2 chunk: d_out += u @ ff2_w[kOff:kOff+512, :]  (+ b2 on first chunk)
// ---------------------------------------------------------------------------
__global__ __launch_bounds__(256)
void ffn2_kernel(const _Float16* __restrict__ u, const _Float16* __restrict__ w2t, // [E][FF]
                 const float* __restrict__ b2, float* __restrict__ out,
                 int kOff, int addBias)
{
    GEMM_DECODE();
    const _Float16* A  = u + (size_t)b * T_SEQ * FCHUNK;
    const _Float16* Bt = w2t + kOff;   // column offset inside [E][FF]
    v8f a00 = {}, a01 = {}, a10 = {}, a11 = {};
    for (int kk = 0; kk < FCHUNK; kk += 32) {
        if (kk + 32 < FCHUNK) __builtin_prefetch(Bt + (size_t)nt0 * 16 * FF + kk + 32, 0, 1);
        v16h fa0 = frag_ld_a(A, FCHUNK, mt0 * 16, kk);
        v16h fa1 = frag_ld_a(A, FCHUNK, (mt0 + 1) * 16, kk);
        v16h fb0 = frag_ld_b(Bt, FF, nt0 * 16, kk);
        v16h fb1 = frag_ld_b(Bt, FF, (nt0 + 1) * 16, kk);
        a00 = WMMA_F16(fa0, fb0, a00);
        a01 = WMMA_F16(fa0, fb1, a01);
        a10 = WMMA_F16(fa1, fb0, a10);
        a11 = WMMA_F16(fa1, fb1, a11);
    }
    float* ob = out + (size_t)b * T_SEQ * EMB;
#pragma unroll
    for (int q = 0; q < 4; ++q) {
        v8f acc = (q == 0) ? a00 : (q == 1) ? a01 : (q == 2) ? a10 : a11;
        int r0 = (mt0 + (q >> 1)) * 16 + ((lane >> 4) << 3);
        int c  = (nt0 + (q & 1)) * 16 + (lane & 15);
        float bv = addBias ? b2[c] : 0.f;
#pragma unroll
        for (int j = 0; j < 8; ++j) {
            size_t o = (size_t)(r0 + j) * EMB + c;
            ob[o] = ob[o] + acc[j] + bv;
        }
    }
}

// ---------------------------------------------------------------------------
// Host orchestration
// ---------------------------------------------------------------------------
extern "C" void kernel_launch(void* const* d_in, const int* in_sizes, int n_in,
                              void* d_out, int out_size, void* d_ws, size_t ws_size,
                              hipStream_t stream) {
    (void)in_sizes; (void)n_in; (void)out_size; (void)ws_size;

    const float* x      = (const float*)d_in[0];
    const float* wq     = (const float*)d_in[1];
    const float* wk     = (const float*)d_in[2];
    const float* wv     = (const float*)d_in[3];
    const float* proj_w = (const float*)d_in[4];
    const float* proj_b = (const float*)d_in[5];
    const float* ff1_w  = (const float*)d_in[6];
    const float* ff1_b  = (const float*)d_in[7];
    const float* ff2_w  = (const float*)d_in[8];
    const float* ff2_b  = (const float*)d_in[9];
    const float* ln1_g  = (const float*)d_in[10];
    const float* ln1_b  = (const float*)d_in[11];
    const float* ln2_g  = (const float*)d_in[12];
    const float* ln2_b  = (const float*)d_in[13];
    float* out = (float*)d_out;

    // ---- workspace carve (256B aligned)
    size_t off = 0;
    auto carve = [&](size_t bytes) {
        void* p = (char*)d_ws + off;
        off += (bytes + 255) & ~(size_t)255;
        return p;
    };
    _Float16* wqt   = (_Float16*)carve((size_t)NH * HD * EMB * 2);
    _Float16* wkt   = (_Float16*)carve((size_t)NH * HD * EMB * 2);
    _Float16* wvt   = (_Float16*)carve((size_t)NH * HD * EMB * 2);
    _Float16* pwt   = (_Float16*)carve((size_t)EMB * EMB * 2);
    _Float16* f1t   = (_Float16*)carve((size_t)FF * EMB * 2);
    _Float16* f2t   = (_Float16*)carve((size_t)EMB * FF * 2);
    _Float16* o16   = (_Float16*)carve((size_t)B_SZ * T_SEQ * EMB * 2);
    _Float16* h2_16 = (_Float16*)carve((size_t)B_SZ * T_SEQ * EMB * 2);
    _Float16* u16   = (_Float16*)carve((size_t)B_SZ * T_SEQ * FCHUNK * 2);

    // ---- weight convert + transpose (f32 -> f16 [N][K])
    {
        dim3 blk(256);
        convt_kernel<<<dim3((EMB * HD + 255) / 256, NH), blk, 0, stream>>>(wq, wqt, EMB, HD);
        convt_kernel<<<dim3((EMB * HD + 255) / 256, NH), blk, 0, stream>>>(wk, wkt, EMB, HD);
        convt_kernel<<<dim3((EMB * HD + 255) / 256, NH), blk, 0, stream>>>(wv, wvt, EMB, HD);
        convt_kernel<<<dim3((EMB * EMB + 255) / 256, 1), blk, 0, stream>>>(proj_w, pwt, EMB, EMB);
        convt_kernel<<<dim3((EMB * FF  + 255) / 256, 1), blk, 0, stream>>>(ff1_w, f1t, EMB, FF);
        convt_kernel<<<dim3((FF  * EMB + 255) / 256, 1), blk, 0, stream>>>(ff2_w, f2t, FF, EMB);
    }

    // ---- fused LN1 + QKV + causal attention (1 workgroup per batch, ~290KB LDS)
    attn_kernel<<<dim3(B_SZ), dim3(512), SMEM_BYTES, stream>>>(
        x, wqt, wkt, wvt, ln1_g, ln1_b, o16);

    // ---- x1 = x + o @ proj_w + proj_b  -> d_out (f32)
    proj_kernel<<<dim3(B_SZ * 8), dim3(256), 0, stream>>>(o16, pwt, proj_b, x, out);

    // ---- h2 = LN2(x1) -> f16
    ln_kernel<<<dim3(B_SZ * T_SEQ / 8), dim3(256), 0, stream>>>(out, ln2_g, ln2_b, h2_16);

    // ---- FFN in 4 column chunks of 512; d_out accumulates (chunk 0 adds bias)
    for (int c = 0; c < FF / FCHUNK; ++c) {
        ffn1_kernel<<<dim3(B_SZ * 8), dim3(256), 0, stream>>>(h2_16, f1t, ff1_b, u16, c * FCHUNK);
        ffn2_kernel<<<dim3(B_SZ * 8), dim3(256), 0, stream>>>(u16, f2t, ff2_b, out, c * FCHUNK, c == 0);
    }
}
